// GCN_4561255269294
// MI455X (gfx1250) — matrix operand order repaired and checked by reading, not compiled, not run on hardware
//
#include <hip/hip_runtime.h>
#include <hip/hip_bf16.h>

// ---------------------------------------------------------------------------
// GCN forward (PyG GCNConv x3 + mean-pool + linear + log_softmax) for gfx1250.
//  - 128x128 GEMMs: v_wmma_f32_16x16x32_f16, B pre-packed to fragment layout,
//    80 rows/block => 20 WMMAs per B-register load.
//  - finalize (dinv*Q + b, ReLU) fused into GEMM LDS staging / pool kernel.
//  - propagation: f32 atomics, working set < 192MB L2.
// ---------------------------------------------------------------------------

typedef __attribute__((ext_vector_type(16))) _Float16 v16h;
typedef __attribute__((ext_vector_type(8)))  _Float16 v8h;
typedef __attribute__((ext_vector_type(4)))  _Float16 v4h;
typedef __attribute__((ext_vector_type(8)))  float    v8f;

#define DIMH 128
#define ROWS 80          // rows per GEMM block: 50000 = 625 * 80, 80 = 5 * 16

// ---- small utility kernels -------------------------------------------------

__global__ void gcn_fill_kernel(float* __restrict__ p, float val, int n) {
    int i = blockIdx.x * blockDim.x + threadIdx.x;
    if (i < n) p[i] = val;
}

// deg[v] = 1 (self loop, pre-filled) + #{edges with dst == v}
__global__ void gcn_deg_kernel(const int* __restrict__ ei, float* __restrict__ deg, int E) {
    int e = blockIdx.x * blockDim.x + threadIdx.x;
    if (e < E) atomicAdd(&deg[ei[E + e]], 1.0f);
}

__global__ void gcn_rsqrt_kernel(float* __restrict__ d, int n) {
    int i = blockIdx.x * blockDim.x + threadIdx.x;
    if (i < n) d[i] = rsqrtf(d[i]);   // deg >= 1 always (self loop)
}

// ---- pack W [128,128] f32 row-major into per-lane f16 B-fragment layout ----
// Index: ((ntile*4 + chunk)*32 + lane) * 16 halves.
// Fragment (B 32x16, this code's contiguous k order, same as A): N = lane&15, K = (lane>>4)*16 + i.

__global__ void gcn_pack_w_kernel(const float* __restrict__ Wt, _Float16* __restrict__ Wpk) {
    int t = blockIdx.x * blockDim.x + threadIdx.x;
    if (t >= 8 * 4 * 32) return;
    int lane  = t & 31;
    int chunk = (t >> 5) & 3;
    int ntile = t >> 7;
    int hlf   = lane >> 4;
    int col   = ntile * 16 + (lane & 15);
    int k0    = chunk * 32;
    _Float16* dst = Wpk + (size_t)t * 16;
    #pragma unroll
    for (int i = 0; i < 16; ++i)
        dst[i] = (_Float16)Wt[(size_t)(k0 + hlf * 16 + i) * DIMH + col];
}

// ---- layer 1: [N,3] @ [3,128], scaled by dinv[row]; writes P and Q ---------

__global__ void gcn_layer1_kernel(const float* __restrict__ x,
                                  const float* __restrict__ W1,
                                  const float* __restrict__ dinv,
                                  float* __restrict__ P, float* __restrict__ Q,
                                  int total) {
    int idx = blockIdx.x * blockDim.x + threadIdx.x;
    if (idx >= total) return;
    int v = idx >> 7;
    int f = idx & (DIMH - 1);
    float h = x[v * 3 + 0] * W1[f]
            + x[v * 3 + 1] * W1[DIMH + f]
            + x[v * 3 + 2] * W1[2 * DIMH + f];
    float val = h * dinv[v];
    P[idx] = val;
    Q[idx] = val;
}

// ---- fused GEMM: a = relu(dinv*Qin + bias); Hs = (a @ W) * dinv ------------
// Writes P (gather source) and Qout (accumulator init; may alias Qin — each
// block reads only its own ROWS rows before the barrier, then rewrites them).
// Grid must be exactly N/ROWS blocks; all 256 lanes active (EXEC all-ones).

__global__ __launch_bounds__(256)
void gcn_gemm_fused_kernel(const float* __restrict__ Qin,
                           const _Float16* __restrict__ Wpk,   // packed fragments
                           const float* __restrict__ dinv,
                           const float* __restrict__ bias,
                           float* __restrict__ P, float* __restrict__ Qout) {
    __shared__ __align__(16) _Float16 ldsA[ROWS * DIMH];   // 20KB f16 A-tile
    __shared__ float ldsDinv[ROWS];

    const int tid = threadIdx.x;
    const int m0  = blockIdx.x * ROWS;

    if (tid < ROWS) ldsDinv[tid] = dinv[m0 + tid];

    // Stage: h = relu(dinv[v]*Q + b) converted to f16, 4 elements per iter.
    for (int i = tid * 4; i < ROWS * DIMH; i += 256 * 4) {
        int row = i >> 7;
        int f   = i & (DIMH - 1);
        float dv = dinv[m0 + row];
        const float4 q = *(const float4*)(Qin + (size_t)(m0 + row) * DIMH + f);
        v4h hv;
        hv[0] = (_Float16)fmaxf(dv * q.x + bias[f + 0], 0.0f);
        hv[1] = (_Float16)fmaxf(dv * q.y + bias[f + 1], 0.0f);
        hv[2] = (_Float16)fmaxf(dv * q.z + bias[f + 2], 0.0f);
        hv[3] = (_Float16)fmaxf(dv * q.w + bias[f + 3], 0.0f);
        *(v4h*)(&ldsA[i]) = hv;
    }
    __syncthreads();

    const int wave = tid >> 5;
    const int lane = tid & 31;
    const int hlf  = lane >> 4;
    const int lrow = lane & 15;
    const int col  = wave * 16 + lrow;

    // B fragments: one register load per K-chunk, reused for all 5 M-subtiles.
    v16h bfrag[4];
    #pragma unroll
    for (int c = 0; c < 4; ++c)
        bfrag[c] = *(const v16h*)(Wpk + (size_t)((wave * 4 + c) * 32 + lane) * 16);

    #pragma unroll
    for (int mt = 0; mt < ROWS / 16; ++mt) {
        const _Float16* Abase = &ldsA[(mt * 16 + lrow) * DIMH];
        v8f acc = {};
        #pragma unroll
        for (int c = 0; c < 4; ++c) {
            const int k0 = c * 32;
            // A fragment: row = lane&15; K = hlf*16 + 0..15
            v8h lo = *(const v8h*)(Abase + k0 + hlf * 16);
            v8h hi = *(const v8h*)(Abase + k0 + hlf * 16 + 8);
            v16h a;
            #pragma unroll
            for (int i = 0; i < 8; ++i) { a[i] = lo[i]; a[i + 8] = hi[i]; }
            acc = __builtin_amdgcn_wmma_f32_16x16x32_f16(
                false, a, false, bfrag[c], (short)0, acc, false, false);
        }
        // D row = mt*16 + hlf*8 + r, col = lane&15; fuse dinv scaling.
        #pragma unroll
        for (int r = 0; r < 8; ++r) {
            int lr  = mt * 16 + hlf * 8 + r;
            float val = acc[r] * ldsDinv[lr];
            size_t off = (size_t)(m0 + lr) * DIMH + col;
            P[off]    = val;   // gather source
            Qout[off] = val;   // accumulator init (self-loop term / dinv[v])
        }
    }
}

// ---- propagation: Q[dst] += P[src] over all edges (f32 atomics to L2) ------

__global__ void gcn_edge_kernel(const int* __restrict__ ei,
                                const float* __restrict__ P,
                                float* __restrict__ Q, int E) {
    int e = blockIdx.x * 8 + (threadIdx.x >> 5);   // one wave per edge
    if (e >= E) return;
    int lane = threadIdx.x & 31;
    int s = ei[e];
    int d = ei[E + e];
    const float4 v = *(const float4*)(P + (size_t)s * DIMH + lane * 4);
    float* q = Q + (size_t)d * DIMH + lane * 4;
    atomicAdd(q + 0, v.x);
    atomicAdd(q + 1, v.y);
    atomicAdd(q + 2, v.z);
    atomicAdd(q + 3, v.w);
}

// ---- fused layer-3 finalize + global mean pool accumulate ------------------
// h3[v] = dinv[v]*Q[v] + b3 (no relu); sums[batch[v]] += h3; cnts[g] += 1.

__global__ void gcn_pool_fused_kernel(const float* __restrict__ Q,
                                      const float* __restrict__ dinv,
                                      const float* __restrict__ b3,
                                      const int* __restrict__ batch,
                                      float* __restrict__ sums,
                                      float* __restrict__ cnts, int N) {
    int v = blockIdx.x * 8 + (threadIdx.x >> 5);   // one wave per node
    if (v >= N) return;
    int lane = threadIdx.x & 31;
    int f = lane * 4;
    float dv = dinv[v];
    const float4 q = *(const float4*)(Q + (size_t)v * DIMH + f);
    int g = batch[v];
    float* s = sums + (size_t)g * DIMH + f;
    atomicAdd(s + 0, dv * q.x + b3[f + 0]);
    atomicAdd(s + 1, dv * q.y + b3[f + 1]);
    atomicAdd(s + 2, dv * q.z + b3[f + 2]);
    atomicAdd(s + 3, dv * q.w + b3[f + 3]);
    if (lane == 0) atomicAdd(&cnts[g], 1.0f);
}

__global__ void gcn_pool_div_kernel(const float* __restrict__ sums,
                                    const float* __restrict__ cnts,
                                    float* __restrict__ hG, int total) {
    int idx = blockIdx.x * blockDim.x + threadIdx.x;
    if (idx >= total) return;
    int g = idx >> 7;
    hG[idx] = sums[idx] / fmaxf(cnts[g], 1.0f);
}

// ---- classifier head: logits = hG @ Wl + bl ; log_softmax ------------------

__global__ void gcn_head_kernel(const float* __restrict__ hG,
                                const float* __restrict__ Wl,
                                const float* __restrict__ bl,
                                float* __restrict__ out, int G) {
    int g = blockIdx.x * blockDim.x + threadIdx.x;
    if (g >= G) return;
    float l0 = bl[0], l1 = bl[1];
    #pragma unroll 4
    for (int k = 0; k < DIMH; ++k) {
        float hv = hG[(size_t)g * DIMH + k];
        l0 += hv * Wl[k * 2 + 0];
        l1 += hv * Wl[k * 2 + 1];
    }
    float m = fmaxf(l0, l1);
    float lse = m + logf(__expf(l0 - m) + __expf(l1 - m));
    out[g * 2 + 0] = l0 - lse;
    out[g * 2 + 1] = l1 - lse;
}

// ---------------------------------------------------------------------------

extern "C" void kernel_launch(void* const* d_in, const int* in_sizes, int n_in,
                              void* d_out, int out_size, void* d_ws, size_t ws_size,
                              hipStream_t stream) {
    const float* x   = (const float*)d_in[0];
    const int*   ei  = (const int*)  d_in[1];   // [2,E]
    const int*   bat = (const int*)  d_in[2];
    const float* W1  = (const float*)d_in[3];
    const float* b1  = (const float*)d_in[4];
    const float* W2  = (const float*)d_in[5];
    const float* b2  = (const float*)d_in[6];
    const float* W3  = (const float*)d_in[7];
    const float* b3  = (const float*)d_in[8];
    const float* Wl  = (const float*)d_in[9];
    const float* bl  = (const float*)d_in[10];

    const int N = in_sizes[0] / 3;        // 50000 (multiple of ROWS=80)
    const int E = in_sizes[1] / 2;        // 600000
    const int G = out_size / (DIMH + 2);  // 512

    // workspace (floats): dinv | B1(P) | B2(Q) | sums | cnts | Wpk2 | Wpk3
    float* ws   = (float*)d_ws;
    const size_t nmat = (size_t)N * DIMH;
    float* dinv = ws;
    float* B1   = dinv + ((N + 63) & ~63);
    float* B2   = B1 + nmat;
    float* sums = B2 + nmat;
    float* cnts = sums + (size_t)G * DIMH;
    _Float16* Wpk2 = (_Float16*)(cnts + ((G + 63) & ~63));
    _Float16* Wpk3 = Wpk2 + 8 * 4 * 32 * 16;

    float* hG   = (float*)d_out;             // [G,128]
    float* logp = hG + (size_t)G * DIMH;     // [G,2]

    const int total = N * DIMH;
    const int tb    = (total + 255) / 256;

    // normalization coefficients + packed weights + pool zeroing
    gcn_fill_kernel<<<(N + 255) / 256, 256, 0, stream>>>(dinv, 1.0f, N);
    gcn_fill_kernel<<<(G * DIMH + G + 255) / 256, 256, 0, stream>>>(sums, 0.0f, G * DIMH + G);
    gcn_deg_kernel<<<(E + 255) / 256, 256, 0, stream>>>(ei, dinv, E);
    gcn_rsqrt_kernel<<<(N + 255) / 256, 256, 0, stream>>>(dinv, N);
    gcn_pack_w_kernel<<<4, 256, 0, stream>>>(W2, Wpk2);
    gcn_pack_w_kernel<<<4, 256, 0, stream>>>(W3, Wpk3);

    // layer 1: 3->128 (VALU), then propagate
    gcn_layer1_kernel<<<tb, 256, 0, stream>>>(x, W1, dinv, B1, B2, total);
    gcn_edge_kernel<<<(E + 7) / 8, 256, 0, stream>>>(ei, B1, B2, E);

    // layer 2: relu(dinv*Q+b1) fused into WMMA GEMM, then propagate
    gcn_gemm_fused_kernel<<<N / ROWS, 256, 0, stream>>>(B2, Wpk2, dinv, b1, B1, B2);
    gcn_edge_kernel<<<(E + 7) / 8, 256, 0, stream>>>(ei, B1, B2, E);

    // layer 3: relu(dinv*Q+b2) fused into WMMA GEMM, then propagate
    gcn_gemm_fused_kernel<<<N / ROWS, 256, 0, stream>>>(B2, Wpk3, dinv, b2, B1, B2);
    gcn_edge_kernel<<<(E + 7) / 8, 256, 0, stream>>>(ei, B1, B2, E);

    // finalize layer 3 (dinv*Q+b3, no relu) fused into mean-pool accumulate
    gcn_pool_fused_kernel<<<(N + 7) / 8, 256, 0, stream>>>(B2, dinv, b3, bat, sums, cnts, N);
    gcn_pool_div_kernel<<<(G * DIMH + 255) / 256, 256, 0, stream>>>(sums, cnts, hG, G * DIMH);
    gcn_head_kernel<<<(G + 255) / 256, 256, 0, stream>>>(hG, Wl, bl, logp, G);
}
